// ParameterKernel_84138409328826
// MI455X (gfx1250) — compile-verified, matrix operand order checked by reading, not run on hardware
//
#include <hip/hip_runtime.h>

// ---------------------------------------------------------------------------
// Fused (X_ref @ X_in^T)^expK * mask -> Alpha @ (.) for MI455X (gfx1250).
// bf16 WMMA (v_wmma_f32_16x16x32_bf16), flash-attention style two-GEMM fusion,
// double-buffered X_ref staging via async global->LDS DMA (ASYNCcnt).
// ---------------------------------------------------------------------------

typedef __attribute__((ext_vector_type(16))) __bf16        v16bf;
typedef __attribute__((ext_vector_type(8)))  float         v8f;
typedef __attribute__((ext_vector_type(4)))  unsigned int  u32x4;

union Frag {
  v16bf          v;      // WMMA A/B operand (16 bf16 = 32 B)
  u32x4          q[2];   // two 16-byte halves for b128 loads
  unsigned short u[16];  // raw bf16 element access
};
union Z8 {
  u32x4 q[2];
  int   z[8];
};

#define N_REF   8192
#define N_IN    8192
#define D_FEAT  384
#define N_PROPS 64
#define J_TILE  128          // query columns per workgroup
#define I_TILE  32           // reference rows per inner iteration
#define N_SPLIT 8            // split-K over i (combined with f32 atomics)
#define I_CHUNK (N_REF / N_SPLIT)   // 1024
#define N_TILES (I_CHUNK / I_TILE)  // 32

// LDS layout (bytes); sXr is double-buffered. Only dynamic LDS is used, so
// these static offsets are also the LDS byte addresses for async DMA.
#define SXI_HALFS (J_TILE * D_FEAT)          // 49152 bf16 = 98304 B
#define SXR_HALFS (I_TILE * D_FEAT)          // 12288 bf16 = 24576 B each
#define SXR_OFF_B (SXI_HALFS * 2)                      // 98304
#define SZR_OFF_B ((SXI_HALFS + 2 * SXR_HALFS) * 2)    // 147456 (16B aligned)
#define SZQ_OFF_B (SZR_OFF_B + I_CHUNK * 4)            // 151552
#define LDS_BYTES (SZQ_OFF_B + J_TILE * 4)             // 152064 (2 WG/WGP)

#define XR_VECS (SXR_HALFS / 8 / 256)        // 6 b128 per thread per tile

template <int N> struct IC { static constexpr int value = N; };

__device__ __forceinline__ unsigned short f2bf(float f) {
  unsigned int u = __float_as_uint(f);
  u += 0x7FFFu + ((u >> 16) & 1u);               // round-to-nearest-even
  return (unsigned short)(u >> 16);
}

__device__ __forceinline__ v8f wmma_bf16(const Frag a, const Frag b, v8f c) {
  // (neg_a, A, neg_b, B, c_mod, C, reuse_a, reuse_b)
  return __builtin_amdgcn_wmma_f32_16x16x32_bf16(false, a.v, false, b.v,
                                                 (short)0, c, false, false);
}

// Async DMA: copy 16 B from global memory straight into LDS, no VGPR transit.
// lds_off is the byte offset within this workgroup's LDS allocation.
__device__ __forceinline__ void async_ld_b128(unsigned lds_off,
                                              const void* gaddr) {
  asm volatile("global_load_async_to_lds_b128 %0, %1, off"
               :: "v"(lds_off), "v"(gaddr)
               : "memory");
}
__device__ __forceinline__ void wait_async0() {
  asm volatile("s_wait_asynccnt 0x0" ::: "memory");
}

// ---------------------------------------------------------------------------
// Pre-pass: fp32 -> bf16 for X_ref / X_in / Alpha into workspace; zero d_out.
// ---------------------------------------------------------------------------
__global__ void cvt_init_kernel(const float* __restrict__ Alpha,
                                const float* __restrict__ Xr,
                                const float* __restrict__ Xi,
                                unsigned short* __restrict__ wAl,
                                unsigned short* __restrict__ wXr,
                                unsigned short* __restrict__ wXi,
                                float* __restrict__ out) {
  const int NXR = N_REF * D_FEAT;                // 3145728
  const int NXI = N_IN * D_FEAT;                 // 3145728
  const int NAL = N_PROPS * N_REF;               // 524288
  const int NOUT = N_PROPS * N_IN;               // 524288
  const int total = NXR + NXI + NAL + NOUT;
  int idx = blockIdx.x * blockDim.x + threadIdx.x;
  const int stride = gridDim.x * blockDim.x;
  for (; idx < total; idx += stride) {
    if (idx < NXR) {
      wXr[idx] = f2bf(Xr[idx]);
    } else if (idx < NXR + NXI) {
      int k = idx - NXR;
      wXi[k] = f2bf(Xi[k]);
    } else if (idx < NXR + NXI + NAL) {
      int k = idx - NXR - NXI;
      wAl[k] = f2bf(Alpha[k]);
    } else {
      out[idx - NXR - NXI - NAL] = 0.0f;
    }
  }
}

// ---------------------------------------------------------------------------
// Fused main kernel.
//   grid.x = N_IN / J_TILE (64 column blocks), grid.y = N_SPLIT (i chunks)
//   256 threads = 8 waves; wave w owns j columns [16w,16w+16) and all 64 props.
// ---------------------------------------------------------------------------
__global__ __launch_bounds__(256, 4)
void fused_poly_kernel(const unsigned short* __restrict__ wXr,
                       const unsigned short* __restrict__ wXi,
                       const unsigned short* __restrict__ wAl,
                       const int* __restrict__ Zr,
                       const int* __restrict__ Zq,
                       const int* __restrict__ expKp,
                       float* __restrict__ out) {
  extern __shared__ __align__(16) char smem[];
  unsigned short* sXi     = (unsigned short*)smem;       // [J_TILE][D_FEAT]
  unsigned short* sXrBase = sXi + SXI_HALFS;             // 2 x [I_TILE][D_FEAT]
  int* sZr = (int*)(smem + SZR_OFF_B);                   // [I_CHUNK]
  int* sZq = (int*)(smem + SZQ_OFF_B);                   // [J_TILE]

  const int tid  = threadIdx.x;
  const int wave = tid >> 5;
  const int lane = tid & 31;
  const int h    = lane >> 4;   // lane half (K offset selector)
  const int n    = lane & 15;   // row (A) / column (B,C,D) within fragment

  const int j0    = blockIdx.x * J_TILE;
  const int iBase = blockIdx.y * I_CHUNK;
  const int expK  = *expKp;

  // ---- prologue: async-DMA X_in tile + X_ref tile 0, load Z tiles ----
  {
    const char* srcXi = (const char*)(wXi + (size_t)j0 * D_FEAT);
    for (int e = tid * 16; e < SXI_HALFS * 2; e += 256 * 16)
      async_ld_b128((unsigned)e, srcXi + e);
    const char* srcXr = (const char*)(wXr + (size_t)iBase * D_FEAT);
#pragma unroll
    for (int v = 0; v < XR_VECS; ++v) {
      const int e = (tid + v * 256) * 16;
      async_ld_b128(SXR_OFF_B + e, srcXr + e);
    }
    for (int idx = tid; idx < J_TILE; idx += 256) sZq[idx] = Zq[j0 + idx];
    for (int idx = tid; idx < I_CHUNK; idx += 256) sZr[idx] = Zr[iBase + idx];
  }
  wait_async0();
  __syncthreads();

  const int zq = sZq[16 * wave + n];        // element label of this lane's column

  const unsigned short* rowB = sXi + (16 * wave + n) * D_FEAT;   // column j
  const u32x4* zrv = (const u32x4*)sZr;

  // Main loop, specialized on the (kernel-uniform) polynomial exponent so the
  // per-element power is straight-line code, not 16 tiny runtime loops.
  auto run = [&](auto ek) {
    constexpr int EK = decltype(ek)::value;

    v8f yacc[4];
#pragma unroll
    for (int pt = 0; pt < 4; ++pt) yacc[pt] = (v8f)0.0f;

    for (int it = 0; it < N_TILES; ++it) {
      const int i0   = iBase + it * I_TILE;   // global i of tile start
      const int iloc = it * I_TILE;           // offset into sZr
      const bool haveNext = (it + 1 < N_TILES);

      // ---- async-DMA next tile into the alternate LDS buffer; the copy is
      //      in flight during both GEMM stages (no VGPR transit at all) ----
      if (haveNext) {
        const char* src = (const char*)(wXr + (size_t)(i0 + I_TILE) * D_FEAT);
        const unsigned ldsBase = SXR_OFF_B + ((it + 1) & 1) * (SXR_HALFS * 2);
#pragma unroll
        for (int v = 0; v < XR_VECS; ++v) {
          const int e = (tid + v * 256) * 16;
          async_ld_b128(ldsBase + e, src + e);
        }
        if (it + 2 < N_TILES && tid < 192)    // warm L2 two tiles ahead
          __builtin_prefetch(
              (const char*)(wXr + (size_t)(i0 + 2 * I_TILE) * D_FEAT) +
                  tid * 128, 0, 0);
      }

      // ---- stage A: S[32 x 128] tile, this wave computes 2 frags (m=0,1).
      //      sched_barrier every 2 k-steps bounds the scheduler's look-ahead
      //      (~2x24 frag VGPRs in flight instead of 12x24 => no spills). ----
      const unsigned short* cur = sXrBase + (it & 1) * SXR_HALFS;
      const unsigned short* rowA0 = cur + n * D_FEAT;
      const unsigned short* rowA1 = cur + (16 + n) * D_FEAT;
      v8f s0 = (v8f)0.0f, s1 = (v8f)0.0f;
#pragma unroll
      for (int ks = 0; ks < D_FEAT / 32; ++ks) {
        const int k0 = 32 * ks;
        Frag a0, a1, b;
        b.q[0]  = ((const u32x4*)rowB)[(k0 + 8 * h) >> 3];
        b.q[1]  = ((const u32x4*)rowB)[(k0 + 16 + 8 * h) >> 3];
        a0.q[0] = ((const u32x4*)rowA0)[(k0 + 8 * h) >> 3];
        a0.q[1] = ((const u32x4*)rowA0)[(k0 + 16 + 8 * h) >> 3];
        a1.q[0] = ((const u32x4*)rowA1)[(k0 + 8 * h) >> 3];
        a1.q[1] = ((const u32x4*)rowA1)[(k0 + 16 + 8 * h) >> 3];
        s0 = wmma_bf16(a0, b, s0);
        s1 = wmma_bf16(a1, b, s1);
        if ((ks & 1) == 1) __builtin_amdgcn_sched_barrier(0);
      }
      __builtin_amdgcn_sched_barrier(0);   // keep aAl loads out of stage A

      // ---- preload all stage-B Alpha A-fragments (one load clause; latency
      //      hides under the mask/convert VALU work below) ----
      Frag aAl[4];
#pragma unroll
      for (int pt = 0; pt < 4; ++pt) {
        const u32x4* rowAl =
            (const u32x4*)(wAl + (size_t)(16 * pt + n) * N_REF);
        aAl[pt].q[0] = rowAl[(i0 + 8 * h) >> 3];
        aAl[pt].q[1] = rowAl[(i0 + 16 + 8 * h) >> 3];
      }

      // ---- Z_ref labels for this lane's 16 i-rows: 4x ds_load_b128 ----
      Z8 z0, z1;
      z0.q[0] = zrv[(iloc + 8 * h) >> 2];
      z0.q[1] = zrv[((iloc + 8 * h) >> 2) + 1];
      z1.q[0] = zrv[(iloc + 16 + 8 * h) >> 2];
      z1.q[1] = zrv[((iloc + 16 + 8 * h) >> 2) + 1];

      // ---- mask + K^expK, convert D frags -> B frag of GEMM2 ----
      // D frag m: element r holds (i = 16m + r + 8h); B frag element e needs
      // K = e + 8h (e<8) or 16 + (e-8) + 8h  => direct concat, no shuffles.
      Frag p;
#pragma unroll
      for (int r = 0; r < 8; ++r) {
        float v = s0[r];
        float pw;
        if constexpr (EK > 0) {
          pw = v;
#pragma unroll
          for (int t2 = 1; t2 < EK; ++t2) pw *= v;
        } else {
          pw = 1.0f;
          for (int t2 = 0; t2 < expK; ++t2) pw *= v;
        }
        p.u[r] = f2bf((z0.z[r] == zq) ? pw : 0.0f);
      }
#pragma unroll
      for (int r = 0; r < 8; ++r) {
        float v = s1[r];
        float pw;
        if constexpr (EK > 0) {
          pw = v;
#pragma unroll
          for (int t2 = 1; t2 < EK; ++t2) pw *= v;
        } else {
          pw = 1.0f;
          for (int t2 = 0; t2 < expK; ++t2) pw *= v;
        }
        p.u[8 + r] = f2bf((z1.z[r] == zq) ? pw : 0.0f);
      }

      // ---- stage B: y[64 x 16(cols of this wave)] += Alpha_tile @ P ----
#pragma unroll
      for (int pt = 0; pt < 4; ++pt)
        yacc[pt] = wmma_bf16(aAl[pt], p, yacc[pt]);

      // ---- flip buffers: DMA done + all waves past their reads ----
      wait_async0();
      __syncthreads();
    }

    // ---- epilogue: combine split-K partials via f32 global atomics ----
    const int col = j0 + 16 * wave + n;
#pragma unroll
    for (int pt = 0; pt < 4; ++pt) {
#pragma unroll
      for (int r = 0; r < 8; ++r) {
        const int prow = 16 * pt + r + 8 * h;
        atomicAdd(&out[prow * N_IN + col], yacc[pt][r]);
      }
    }
  };

  if (expK == 2) run(IC<2>{});      // fast path used by this problem
  else           run(IC<0>{});      // generic runtime exponent
}

// ---------------------------------------------------------------------------
// Host launcher
// ---------------------------------------------------------------------------
extern "C" void kernel_launch(void* const* d_in, const int* in_sizes, int n_in,
                              void* d_out, int out_size, void* d_ws, size_t ws_size,
                              hipStream_t stream) {
  const float* Alpha = (const float*)d_in[0];   // [64, 8192]
  const float* X_ref = (const float*)d_in[1];   // [8192, 384]
  const float* desc  = (const float*)d_in[2];   // [128, 64, 384] -> [8192, 384]
  const int*   Z_ref = (const int*)d_in[3];     // [8192]
  const int*   Z     = (const int*)d_in[4];     // [8192]
  const int*   expK  = (const int*)d_in[5];     // scalar
  float* out = (float*)d_out;                   // [64, 8192]

  unsigned short* ws  = (unsigned short*)d_ws;
  unsigned short* wXr = ws;                                    // bf16 X_ref
  unsigned short* wXi = wXr + (size_t)N_REF * D_FEAT;          // bf16 X_in
  unsigned short* wAl = wXi + (size_t)N_IN * D_FEAT;           // bf16 Alpha

  cvt_init_kernel<<<8192, 256, 0, stream>>>(Alpha, X_ref, desc,
                                            wAl, wXr, wXi, out);

  dim3 grid(N_IN / J_TILE, N_SPLIT);   // 64 x 8 = 512 workgroups
  fused_poly_kernel<<<grid, dim3(256), (size_t)LDS_BYTES, stream>>>(
      wXr, wXi, wAl, Z_ref, Z, expK, out);
}
